// DSQGAttentionV3_48172353192381
// MI455X (gfx1250) — matrix operand order repaired and checked by reading, not compiled, not run on hardware
//
#include <hip/hip_runtime.h>

typedef __attribute__((ext_vector_type(16))) _Float16 v16h;
typedef __attribute__((ext_vector_type(8)))  float    v8f;

#define H_   16
#define N_   4096
#define HD_  64
#define ND_  44

// One wave (32 threads) handles one head and a 16-query tile.
// Phase 1: banded QK^T via WMMA + Q*se^T GEMM via WMMA; tiles stored whole to LDS.
// Phase 2: masked softmax over 44 deltas, split across the two half-waves
//          (lane%16 = query, lane/16 = delta half), shuffle-combined.
// Phase 3: banded alpha*V via WMMA (alpha fragments built by clamp+select).
__global__ __launch_bounds__(32)
void dsqg_attn_kernel(const float* __restrict__ q,
                      const float* __restrict__ k,
                      const float* __restrict__ v,
                      const float* __restrict__ pb,
                      const float* __restrict__ se,
                      float* __restrict__ out)
{
    static constexpr int EXTD[3] = {128, 384, 1536};

    const int lane = threadIdx.x;      // 0..31
    const int n0   = blockIdx.x * 16;  // query tile base
    const int h    = blockIdx.y;
    const int lc   = lane & 15;
    const bool hi  = lane >= 16;

    __shared__ float sP[7][16][17];               // raw QK^T tiles (4 band + 3 extras)
    __shared__ float sSE[16][48];                 // q . se per (query, delta-idx)
    __shared__ float aE[16][48];                  // exp(score - max) per (query, delta-idx)
    __shared__ float pbL[48];
    __shared__ __align__(16) float vL[112][HD_];  // staged V rows (band + 3 extras)

    const float* qh = q + (size_t)h * N_ * HD_;
    const float* kh = k + (size_t)h * N_ * HD_;
    const float* vh = v + (size_t)h * N_ * HD_;

    // ---- stage pos_bias column for this head ----
    for (int i = lane; i < ND_; i += 32) pbL[i] = pb[i * H_ + h];

    // ---- stage V rows to LDS (coalesced float4) ----
    // rows 0..63 : j = n0-48+r   (band)
    // rows 64..79: j = n0-128+(r-64), 80..95: n0-384, 96..111: n0-1536
    for (int t = lane; t < 112 * (HD_ / 4); t += 32) {
        int r  = t >> 4;
        int c4 = (t & 15) << 2;
        int j;
        if (r < 64)      j = n0 - 48   + r;
        else if (r < 80) j = n0 - 128  + (r - 64);
        else if (r < 96) j = n0 - 384  + (r - 80);
        else             j = n0 - 1536 + (r - 96);
        if (j < 0) j = 0;                       // clamped rows get alpha==0 anyway
        const float4 val = *(const float4*)(vh + (size_t)j * HD_ + c4);
        *(float4*)(&vL[r][c4]) = val;
    }

    // ---- Q tile as two A fragments (d in [0,32) and [32,64)) ----
    // A layout: M = lane%16 ; K = t + (hi ? 16:0)
    v16h A[2];
    {
        const float* qrow = qh + (size_t)(n0 + lc) * HD_;
        #pragma unroll
        for (int dh = 0; dh < 2; ++dh) {
            #pragma unroll
            for (int t = 0; t < 16; ++t) {
                int d = 32 * dh + (hi ? 16 : 0) + t;
                A[dh][t] = (_Float16)qrow[d];
            }
        }
    }

    // ---- QK^T over 7 key tiles (4 band + 3 extras); store whole tiles ----
    #pragma unroll
    for (int tile = 0; tile < 7; ++tile) {
        const int j0 = (tile < 4) ? (n0 - 48 + 16 * tile) : (n0 - EXTD[tile - 4]);
        int j = j0 + lc; if (j < 0) j = 0;
        const float* krow = kh + (size_t)j * HD_;
        v8f acc = {};
        #pragma unroll
        for (int dh = 0; dh < 2; ++dh) {
            // B layout: N = lane%16 ; K = t + (hi ? 16:0)  -> 16 contiguous floats/lane
            v16h B;
            #pragma unroll
            for (int t = 0; t < 16; ++t)
                B[t] = (_Float16)krow[32 * dh + (hi ? 16 : 0) + t];
            acc = __builtin_amdgcn_wmma_f32_16x16x32_f16(false, A[dh], false, B,
                                                         (short)0, acc, false, false);
        }
        // C layout: M = r + (hi?8:0), N = lane%16 -> one base, immediate offsets
        float* dst = &sP[tile][hi ? 8 : 0][0] + lc;
        #pragma unroll
        for (int r = 0; r < 8; ++r) dst[r * 17] = acc[r];
    }

    // ---- Q * se^T (true GEMM, 3 N-tiles of 16) ----
    #pragma unroll
    for (int st = 0; st < 3; ++st) {
        const int ic = 16 * st + lc;                 // < 48, array is padded
        const int iclamp = ic > 43 ? 43 : ic;
        const float* srow = se + (size_t)iclamp * HD_;
        v8f acc = {};
        #pragma unroll
        for (int dh = 0; dh < 2; ++dh) {
            v16h B;
            #pragma unroll
            for (int t = 0; t < 16; ++t)
                B[t] = (_Float16)srow[32 * dh + (hi ? 16 : 0) + t];
            acc = __builtin_amdgcn_wmma_f32_16x16x32_f16(false, A[dh], false, B,
                                                         (short)0, acc, false, false);
        }
        float* dst = &sSE[hi ? 8 : 0][0] + ic;
        #pragma unroll
        for (int r = 0; r < 8; ++r) dst[r * 48] = acc[r];
    }

    __syncthreads();

    // ---- masked softmax over 44 deltas; lane lc owns query n0+lc,
    //      half-wave `hi` handles delta indices [22,44), other half [0,22) ----
    const int qn = n0 + lc;
    const int ib = hi ? 22 : 0;
    float pmx = -3.0e38f;
    #pragma unroll 1
    for (int i = ib; i < ib + 22; ++i) {
        float raw;
        int d;
        if (i < 41) {
            int idx = lc + 48 - i;               // key-row within band block, in [8,63]
            raw = sP[idx >> 4][lc][idx & 15];
            d = i;
        } else {
            raw = sP[4 + (i - 41)][lc][lc];      // diagonal of extra tile
            d = (i == 41) ? 128 : ((i == 42) ? 384 : 1536);
        }
        float x = (qn >= d) ? (raw + sSE[lc][i]) * 0.125f + pbL[i] : -3.0e38f;
        aE[lc][i] = x;
        pmx = fmaxf(pmx, x);
    }
    const float mx = fmaxf(pmx, __shfl_xor(pmx, 16));
    float psum = 0.0f;
    #pragma unroll 1
    for (int i = ib; i < ib + 22; ++i) {
        float e = __expf(aE[lc][i] - mx);
        psum += e;
        aE[lc][i] = e;
    }
    const float inv = 1.0f / (psum + __shfl_xor(psum, 16));  // delta=0 always valid
    __syncthreads();

    // ---- build alpha A fragments directly (clamp + select, no scatter) ----
    // band block 0 rows c=kk      : delta = lc + 48 - kk  in [17,63], valid if <=40
    // band block 1 rows c=32+kk   : delta = lc + 16 - kk  in [-15,31], valid if >=0
    const float e41 = aE[lc][41] * inv;
    const float e42 = aE[lc][42] * inv;
    const float e43 = aE[lc][43] * inv;
    v16h Aa[4];
    #pragma unroll
    for (int t = 0; t < 16; ++t) {
        int kk = (hi ? 16 : 0) + t;
        int d0 = lc + 48 - kk;
        int d0c = d0 > 47 ? 47 : d0;
        float a0 = (d0 <= 40) ? aE[lc][d0c] * inv : 0.0f;
        int d1 = lc + 16 - kk;
        int d1c = d1 < 0 ? 0 : d1;
        float a1 = (d1 >= 0) ? aE[lc][d1c] * inv : 0.0f;
        float a2 = (kk == lc) ? e41 : ((kk == lc + 16) ? e42 : 0.0f);
        float a3 = (kk == lc) ? e43 : 0.0f;
        Aa[0][t] = (_Float16)a0;
        Aa[1][t] = (_Float16)a1;
        Aa[2][t] = (_Float16)a2;
        Aa[3][t] = (_Float16)a3;
    }

    // ---- alpha * V : 4 HD chunks x 4 K-blocks ----
    #pragma unroll
    for (int hc = 0; hc < 4; ++hc) {
        const int hd = 16 * hc + lc;
        v8f acc = {};
        #pragma unroll
        for (int kb = 0; kb < 4; ++kb) {
            v16h B;
            #pragma unroll
            for (int t = 0; t < 16; ++t) {
                int kk = (hi ? 16 : 0) + t;
                int row;
                if (kb < 2)       row = 32 * kb + kk;      // band rows 0..63
                else if (kb == 2) row = 64 + kk;           // 128-delta then 384-delta rows
                else              row = 96 + (kk & 15);    // 1536 rows (upper half: Aa==0)
                B[t] = (_Float16)vL[row][hd];
            }
            acc = __builtin_amdgcn_wmma_f32_16x16x32_f16(false, Aa[kb], false, B,
                                                         (short)0, acc, false, false);
        }
        #pragma unroll
        for (int r = 0; r < 8; ++r) {
            int M = r + (hi ? 8 : 0);
            out[((size_t)h * N_ + (n0 + M)) * HD_ + hd] = acc[r];
        }
    }
}

extern "C" void kernel_launch(void* const* d_in, const int* in_sizes, int n_in,
                              void* d_out, int out_size, void* d_ws, size_t ws_size,
                              hipStream_t stream) {
    (void)in_sizes; (void)n_in; (void)out_size; (void)d_ws; (void)ws_size;
    const float* q  = (const float*)d_in[0];
    const float* k  = (const float*)d_in[1];
    const float* v  = (const float*)d_in[2];
    const float* pb = (const float*)d_in[3];
    const float* se = (const float*)d_in[4];
    float* out = (float*)d_out;
    dim3 grid(N_ / 16, H_);
    dsqg_attn_kernel<<<grid, 32, 0, stream>>>(q, k, v, pb, se, out);
}
